// TokenMergingLayer_57612691309472
// MI455X (gfx1250) — compile-verified, matrix operand order checked by reading, not run on hardware
//
#include <hip/hip_runtime.h>
#include <hip/hip_bf16.h>

typedef __bf16 v16bf __attribute__((ext_vector_type(16)));
typedef __bf16 v8bf  __attribute__((ext_vector_type(8)));
typedef __bf16 v4bf  __attribute__((ext_vector_type(4)));
typedef float  v8f   __attribute__((ext_vector_type(8)));
typedef float  v4f   __attribute__((ext_vector_type(4)));
typedef unsigned int v4u __attribute__((ext_vector_type(4)));
typedef int    v8i   __attribute__((ext_vector_type(8)));

#define DM     1024          // model dim (W is DM x DM)
#define NROWS  8192          // B*(S-R) == B*R == 8192
#define RPB    2048          // reduce/keep rows per batch
#define BM     128           // block tile M (reduced rows)
#define BN     128           // block tile N (output cols)
#define BK     32            // K step (bf16 WMMA K)
#define LDT    (BK + 8)      // padded LDS row stride: 80B -> conflict-free 16-lane frags
#define LDS_TILE (BM * LDT)  // one tile = 128 rows x 40 bf16 = 10240 B

union BF16x16 { v16bf v; v8bf h[2]; };

static __device__ inline v16bf lds_load16(const __bf16* p) {
  BF16x16 t;
  t.h[0] = *(const v8bf*)(p);
  t.h[1] = *(const v8bf*)(p + 8);
  return t.v;
}

// ---------------------------------------------------------------------------
// TDM: issue a 2D bf16 tile load (32 x 128 rows, row stride DM elements) into
// LDS with hardware padding (64B data + 16B pad per row == LDT stride).
// D# group0/group1 bitfields per CDNA5 ISA.
// 2-group form (VADDR2/3 = NULL) is enough for 2D tensors.
// ---------------------------------------------------------------------------
static __device__ inline void tdm_load_tile(unsigned lds_addr, const __bf16* gsrc) {
  unsigned long long ga = (unsigned long long)(size_t)gsrc;
  v4u g0;
  g0[0] = 1u;                                   // count=1 (valid, user mode)
  g0[1] = lds_addr;                             // lds_addr (bytes, wave-relative)
  g0[2] = (unsigned)(ga & 0xffffffffu);         // global_addr[31:0]
  g0[3] = (unsigned)((ga >> 32) & 0x01ffffffu)  // global_addr[56:32]
        | (2u << 30);                           // type = 2 ("image")
  v8i g1;
  // data_size=1 (2B) | pad_enable | pad_interval=3 (16 DWORDs) | pad_amount=3 (4 DWORDs)
  g1[0] = (1 << 16) | (1 << 20) | (3 << 22) | (3 << 25);
  // tensor_dim0 = DM  (bits 79:48), tensor_dim1 = NROWS (bits 111:80)
  g1[1] = (DM & 0xffff) << 16;                  // tensor_dim0[15:0]
  g1[2] = (DM >> 16) | ((NROWS & 0xffff) << 16);// tensor_dim0[31:16] | tensor_dim1[15:0]
  g1[3] = (NROWS >> 16) | (BK << 16);           // tensor_dim1[31:16] | tile_dim0=32
  g1[4] = BM;                                   // tile_dim1=128 rows (tile_dim2=0)
  g1[5] = DM;                                   // tensor_dim0_stride[31:0] = 1024 elems
  g1[6] = 0;                                    // stride[47:32] | dim1_stride[15:0]
  g1[7] = 0;
  asm volatile("tensor_load_to_lds %0, %1" :: "s"(g0), "s"(g1) : "memory");
}

// ---------------------------------------------------------------------------
// Kernel 1: split W (fp32) into hi/lo bf16 pair for bf16x3 GEMM
// ---------------------------------------------------------------------------
__global__ void split_w_kernel(const float* __restrict__ W,
                               __bf16* __restrict__ Whi,
                               __bf16* __restrict__ Wlo) {
  int i = (blockIdx.x * blockDim.x + threadIdx.x) * 4;
  v4f w = *(const v4f*)(W + i);
  v4bf h, l;
#pragma unroll
  for (int j = 0; j < 4; ++j) {
    __bf16 hj = (__bf16)w[j];
    h[j] = hj;
    l[j] = (__bf16)(w[j] - (float)hj);
  }
  *(v4bf*)(Whi + i) = h;
  *(v4bf*)(Wlo + i) = l;
}

// ---------------------------------------------------------------------------
// Kernel 2: gather reduce-rows of x and split fp32 -> hi/lo bf16 (compact A)
// ---------------------------------------------------------------------------
__global__ void split_a_kernel(const float* __restrict__ x,
                               const int* __restrict__ red,
                               __bf16* __restrict__ Ahi,
                               __bf16* __restrict__ Alo) {
  int n = blockIdx.x;
  int src = red[n];
  int c = threadIdx.x * 4;
  v4f v = *(const v4f*)(x + (size_t)src * DM + c);
  v4bf h, l;
#pragma unroll
  for (int j = 0; j < 4; ++j) {
    __bf16 hj = (__bf16)v[j];
    h[j] = hj;
    l[j] = (__bf16)(v[j] - (float)hj);
  }
  *(v4bf*)(Ahi + (size_t)n * DM + c) = h;
  *(v4bf*)(Alo + (size_t)n * DM + c) = l;
}

// ---------------------------------------------------------------------------
// Kernel 3: dest[n] = position j in keep[] with keep[j] == reduce[n]+1, else -1
// keep[] is sorted ascending within each batch segment -> binary search
// ---------------------------------------------------------------------------
__global__ void make_dest_kernel(const int* __restrict__ keep,
                                 const int* __restrict__ red,
                                 int* __restrict__ dest) {
  int n = blockIdx.x * blockDim.x + threadIdx.x;
  if (n >= NROWS) return;
  int target = red[n] + 1;
  int b = n / RPB;
  const int* seg = keep + b * RPB;
  int lo = 0, hi = RPB;
  while (lo < hi) {
    int mid = (lo + hi) >> 1;
    if (seg[mid] < target) lo = mid + 1; else hi = mid;
  }
  dest[n] = (lo < RPB && seg[lo] == target) ? (b * RPB + lo) : -1;
}

// ---------------------------------------------------------------------------
// Kernel 4: out[j] = x[keep[j]]   (GEMM epilogue later adds on top)
// ---------------------------------------------------------------------------
__global__ void gather_copy_kernel(const float* __restrict__ x,
                                   const int* __restrict__ keep,
                                   float* __restrict__ out) {
  int j = blockIdx.x;
  int src = keep[j];
  int c = threadIdx.x * 4;
  v4f v = *(const v4f*)(x + (size_t)src * DM + c);
  *(v4f*)(out + (size_t)j * DM + c) = v;
}

// ---------------------------------------------------------------------------
// Kernel 5: bf16x3 WMMA GEMM (pre-split bf16 A/B) with TDM-staged, double-
// buffered LDS tiles + guarded scatter-add epilogue.
//   C[n, o] = sum_d A[n, d] * W[o, d]
// Block: 256 threads (8 waves). Block tile 128x128, wave tile 32x64.
// Steady state per wave per K-step: 24 ds_load_b128 + 24 v_wmma + 1 barrier;
// wave 0 additionally issues 4 tensor_load_to_lds for the NEXT K-step, so the
// TDM DMA overlaps the whole block's WMMA compute.
// ---------------------------------------------------------------------------
__global__ __launch_bounds__(256)
void gemm_scatter_kernel(const __bf16* __restrict__ Ahi,
                         const __bf16* __restrict__ Alo,
                         const __bf16* __restrict__ Whi,
                         const __bf16* __restrict__ Wlo,
                         const int* __restrict__ dest,
                         float* __restrict__ out) {
  __shared__ __bf16 sA[2][2][LDS_TILE];   // [buf][hi/lo][row*LDT+k]
  __shared__ __bf16 sB[2][2][LDS_TILE];
  __shared__ int sDst[BM];

  const int tid = threadIdx.x;
  const int m0  = blockIdx.y * BM;
  const int n0  = blockIdx.x * BN;

  if (tid < BM) sDst[tid] = dest[m0 + tid];

  const int lane   = tid & 31;
  const int wave   = tid >> 5;
  const int laneHi = lane >> 4;          // 0: lanes 0-15, 1: lanes 16-31
  const int lcol   = lane & 15;
  const int wm     = (wave >> 1) * 32;   // 0,32,64,96
  const int wn     = (wave & 1) * 64;    // 0,64

  v8f acc[2][4];
#pragma unroll
  for (int mi = 0; mi < 2; ++mi)
#pragma unroll
    for (int ni = 0; ni < 4; ++ni)
      acc[mi][ni] = v8f{0.f, 0.f, 0.f, 0.f, 0.f, 0.f, 0.f, 0.f};

  const __bf16* agh = Ahi + (size_t)m0 * DM;
  const __bf16* agl = Alo + (size_t)m0 * DM;
  const __bf16* bgh = Whi + (size_t)n0 * DM;
  const __bf16* bgl = Wlo + (size_t)n0 * DM;

  // prologue: wave 0 issues the 4 tiles of K-step 0 into buffer 0
  if (tid < 32) {
    tdm_load_tile((unsigned)(size_t)&sA[0][0][0], agh);
    tdm_load_tile((unsigned)(size_t)&sA[0][1][0], agl);
    tdm_load_tile((unsigned)(size_t)&sB[0][0][0], bgh);
    tdm_load_tile((unsigned)(size_t)&sB[0][1][0], bgl);
  }

  for (int k0 = 0; k0 < DM; k0 += BK) {
    const int buf = (k0 >> 5) & 1;
    if (tid < 32) __builtin_amdgcn_s_wait_tensorcnt(0);  // this buf's DMA done
    __syncthreads();                                     // visible to all waves
    if (tid < 32 && (k0 + BK) < DM) {                    // prefetch next K tile
      const int nb = buf ^ 1;
      tdm_load_tile((unsigned)(size_t)&sA[nb][0][0], agh + k0 + BK);
      tdm_load_tile((unsigned)(size_t)&sA[nb][1][0], agl + k0 + BK);
      tdm_load_tile((unsigned)(size_t)&sB[nb][0][0], bgh + k0 + BK);
      tdm_load_tile((unsigned)(size_t)&sB[nb][1][0], bgl + k0 + BK);
    }

    // A frag: lane holds row (base+lcol), 16 contiguous K (half by laneHi)
    v16bf aHi[2], aLo[2];
#pragma unroll
    for (int mi = 0; mi < 2; ++mi) {
      const int mrow = wm + mi * 16 + lcol;
      aHi[mi] = lds_load16(&sA[buf][0][mrow * LDT + laneHi * 16]);
      aLo[mi] = lds_load16(&sA[buf][1][mrow * LDT + laneHi * 16]);
    }
#pragma unroll
    for (int ni = 0; ni < 4; ++ni) {
      const int ocol = wn + ni * 16 + lcol;  // B frag: lane = output column
      v16bf bHi = lds_load16(&sB[buf][0][ocol * LDT + laneHi * 16]);
      v16bf bLo = lds_load16(&sB[buf][1][ocol * LDT + laneHi * 16]);
#pragma unroll
      for (int mi = 0; mi < 2; ++mi) {
        acc[mi][ni] = __builtin_amdgcn_wmma_f32_16x16x32_bf16(
            false, aLo[mi], false, bHi, (short)0, acc[mi][ni], false, false);
        acc[mi][ni] = __builtin_amdgcn_wmma_f32_16x16x32_bf16(
            false, aHi[mi], false, bLo, (short)0, acc[mi][ni], false, false);
        acc[mi][ni] = __builtin_amdgcn_wmma_f32_16x16x32_bf16(
            false, aHi[mi], false, bHi, (short)0, acc[mi][ni], false, false);
      }
    }
    // no trailing barrier: the top-of-loop barrier of the NEXT iteration
    // separates this buffer's last reader from its next TDM overwrite
    // (buffers are reused with a 2-iteration distance).
  }

  // ---- epilogue: out[dest] += acc (each (row,col) owned by exactly one lane)
  // C frag layout: VGPR g -> row g (lanes 0-15) / row g+8 (lanes 16-31)
#pragma unroll
  for (int mi = 0; mi < 2; ++mi) {
#pragma unroll
    for (int g = 0; g < 8; ++g) {
      const int mrow = wm + mi * 16 + laneHi * 8 + g;
      const int j = sDst[mrow];
      if (j >= 0) {
        float* orow = out + (size_t)j * DM + n0 + wn + lcol;
#pragma unroll
        for (int ni = 0; ni < 4; ++ni)
          orow[ni * 16] += acc[mi][ni][g];
      }
    }
  }
}

// ---------------------------------------------------------------------------
extern "C" void kernel_launch(void* const* d_in, const int* in_sizes, int n_in,
                              void* d_out, int out_size, void* d_ws, size_t ws_size,
                              hipStream_t stream) {
  const float* x    = (const float*)d_in[0];
  const float* W    = (const float*)d_in[1];
  const int*   keep = (const int*)d_in[2];
  const int*   red  = (const int*)d_in[3];
  float* out = (float*)d_out;

  // workspace: Whi 2MB | Wlo 2MB | Ahi 16MB | Alo 16MB | dest 32KB  (~36MB)
  char* ws = (char*)d_ws;
  __bf16* Whi  = (__bf16*)(ws);
  __bf16* Wlo  = (__bf16*)(ws + (2u << 20));
  __bf16* Ahi  = (__bf16*)(ws + (4u << 20));
  __bf16* Alo  = (__bf16*)(ws + (20u << 20));
  int*    dest = (int*)  (ws + (36u << 20));

  split_w_kernel<<<(DM * DM / 4 + 255) / 256, 256, 0, stream>>>(W, Whi, Wlo);
  split_a_kernel<<<NROWS, 256, 0, stream>>>(x, red, Ahi, Alo);
  make_dest_kernel<<<(NROWS + 255) / 256, 256, 0, stream>>>(keep, red, dest);
  gather_copy_kernel<<<NROWS, 256, 0, stream>>>(x, keep, out);

  dim3 grid(DM / BN, NROWS / BM);   // 8 x 64 blocks
  gemm_scatter_kernel<<<grid, 256, 0, stream>>>(Ahi, Alo, Whi, Wlo, dest, out);
}
